// AdditiveClassifier_82240033784369
// MI455X (gfx1250) — compile-verified, matrix operand order checked by reading, not run on hardware
//
#include <hip/hip_runtime.h>
#include <hip/hip_bf16.h>
#include <stdint.h>

typedef __bf16 bf16_t;
typedef __attribute__((ext_vector_type(16))) __bf16 v16bf;
typedef __attribute__((ext_vector_type(8)))  __bf16 v8bf;
typedef __attribute__((ext_vector_type(8)))  float  v8f;
typedef __attribute__((ext_vector_type(4)))  float  v4f;

#define D_DIM 1024
#define H_DIM 512
#define C_DIM 2
#define MT    64                 // rows per block
#define KT    32                 // K per wmma step (bf16)
#define NKT   (D_DIM / KT)       // 32 K steps
#define NT    16                 // N-tiles (of 16 cols) per wave: 256 cols
#define AROW  40                 // padded bf16 row stride for A tile (80 bytes)
#define BTILE_ELEMS (H_DIM * KT)             // 16384 bf16
#define BTILE_BYTES (BTILE_ELEMS * 2)        // 32 KB contiguous B tile

// ---------------------------------------------------------------------------
// Prep: W1 (f32 [D][H]) -> bf16, K-tiled layout w1p[kt][h][ki] so that each
// K-step's B tile is one contiguous 32KB block (async-to-LDS friendly).
// Also zeroes the bag-logits region of d_out.
// ---------------------------------------------------------------------------
__global__ void pack_w1_kernel(const float* __restrict__ W1,
                               bf16_t* __restrict__ w1p,
                               float* __restrict__ out_logits, int bc) {
    int o = blockIdx.x * blockDim.x + threadIdx.x;
    if (o < D_DIM * H_DIM) {
        int kt  = o >> 14;        // / (H_DIM*KT) = /16384
        int rem = o & 16383;
        int h   = rem >> 5;       // / KT
        int ki  = rem & 31;
        int k   = kt * KT + ki;
        w1p[o] = (bf16_t)W1[k * H_DIM + h];
    }
    if (o < bc) out_logits[o] = 0.0f;
}

// 16-byte-pair fragment load from LDS (two ds_load_b128)
static __device__ __forceinline__ v16bf load_frag(const bf16_t* p) {
    v8bf lo = *(const v8bf*)(p);
    v8bf hi = *(const v8bf*)(p + 16);
    return __builtin_shufflevector(lo, hi,
        0, 1, 2, 3, 4, 5, 6, 7, 8, 9, 10, 11, 12, 13, 14, 15);
}

// ---------------------------------------------------------------------------
// Fused: attended = att*F ; h = relu(attended@W1 + b1) ; pl = h@W2 + b2 ;
// logits = segment_sum(pl).  bf16 WMMA, f32 accumulate, double-buffered
// async-DMA pipeline (ASYNCcnt) for the W1 tiles.
// Block: 256 threads = 8 waves. Wave w: msub = w&3 (16-row stripe of 64),
// hg = w>>2 (two 256-column H groups, 16 N-tiles each).
// ---------------------------------------------------------------------------
__global__ __launch_bounds__(256) void mil_fused_kernel(
    const float* __restrict__ F, const float* __restrict__ att,
    const int* __restrict__ bag_sizes, const bf16_t* __restrict__ w1p,
    const float* __restrict__ b1, const float* __restrict__ W2,
    const float* __restrict__ b2, float* __restrict__ out,
    int N, int NB) {

    __shared__ bf16_t sA[2 * MT * AROW];      // 2 x 5 KB attention-gated A tiles
    __shared__ bf16_t sB[2 * BTILE_ELEMS];    // 2 x 32 KB W1 tiles (async DMA)
    __shared__ float  sAtt[MT];
    __shared__ float  sPL[MT * C_DIM];

    const int tid  = threadIdx.x;
    const int lane = tid & 31;
    const int wave = tid >> 5;
    const int half = (lane >> 4) & 1;
    const int l16  = lane & 15;
    const int msub = wave & 3;               // 16-row stripe (0..3)
    const int hg   = wave >> 2;              // 256-col H group (0..1)
    const int m0   = blockIdx.x * MT;

    if (tid < MT)         sAtt[tid] = att[m0 + tid];
    if (tid < MT * C_DIM) sPL[tid]  = 0.0f;
    __syncthreads();

    v8f acc[NT];
#pragma unroll
    for (int i = 0; i < NT; ++i) {
        v8f z = {0.f, 0.f, 0.f, 0.f, 0.f, 0.f, 0.f, 0.f};
        acc[i] = z;
    }

    const uint32_t ldsB0 = (uint32_t)(uintptr_t)(&sB[0]);
    const uint32_t voff  = (uint32_t)tid * 16u;
    const uint64_t gB0   = (uint64_t)(uintptr_t)w1p;

    // A staging: thread -> (row 0..63, 8 cols)
    const int r0 = tid >> 2;
    const int c0 = (tid & 3) * 8;
    const float aw = sAtt[r0];
    const float* Frow = &F[(size_t)(m0 + r0) * D_DIM + c0];

    // ---- prologue: tile 0 ----
#pragma unroll
    for (int i = 0; i < 8; ++i) {
        asm volatile("global_load_async_to_lds_b128 %0, %1, %2"
                     :
                     : "v"(ldsB0 + voff + (uint32_t)(i * 4096)),
                       "v"(voff + (uint32_t)(i * 4096)),
                       "s"(gB0)
                     : "memory");
    }
    {
        v4f f0 = *(const v4f*)(Frow + 0);
        v4f f1 = *(const v4f*)(Frow + 4);
        v8bf a;
#pragma unroll
        for (int j = 0; j < 4; ++j) { a[j] = (bf16_t)(f0[j] * aw); a[4 + j] = (bf16_t)(f1[j] * aw); }
        *(v8bf*)(&sA[r0 * AROW + c0]) = a;
    }
    asm volatile("s_wait_asynccnt 0" ::: "memory");
    __syncthreads();

    // ---- main pipelined K loop ----
    for (int kt = 0; kt < NKT; ++kt) {
        const int cur = kt & 1;
        const int nxt = cur ^ 1;

        // issue next tile's DMA + global loads before computing current tile
        v4f f0, f1;
        if (kt + 1 < NKT) {
            const uint64_t gbase = gB0 + (uint64_t)(kt + 1) * BTILE_BYTES;
            const uint32_t ldsb  = ldsB0 + (uint32_t)(nxt * BTILE_BYTES);
#pragma unroll
            for (int i = 0; i < 8; ++i) {
                asm volatile("global_load_async_to_lds_b128 %0, %1, %2"
                             :
                             : "v"(ldsb + voff + (uint32_t)(i * 4096)),
                               "v"(voff + (uint32_t)(i * 4096)),
                               "s"(gbase)
                             : "memory");
            }
            f0 = *(const v4f*)(Frow + (kt + 1) * KT + 0);
            f1 = *(const v4f*)(Frow + (kt + 1) * KT + 4);
        }
        if (kt + 2 < NKT) {
            __builtin_prefetch(Frow + (kt + 2) * KT, 0, 1);   // global_prefetch_b8
        }

        // ---- compute current tile ----
        const bf16_t* aBase = &sA[cur * MT * AROW + (msub * 16 + l16) * AROW + (half ? 8 : 0)];
        const bf16_t* bBase = &sB[cur * BTILE_ELEMS + (hg * 256 + l16) * KT + (half ? 8 : 0)];
        v16bf afrag = load_frag(aBase);

        v16bf bfrag[2];
        bfrag[0] = load_frag(bBase);
#pragma unroll
        for (int nt = 0; nt < NT; ++nt) {
            if (nt + 1 < NT)
                bfrag[(nt + 1) & 1] = load_frag(bBase + (nt + 1) * 16 * KT);
            acc[nt] = __builtin_amdgcn_wmma_f32_16x16x32_bf16(
                false, afrag, false, bfrag[nt & 1],
                (short)0, acc[nt], false, false);
        }

        // stage next A tile into the other buffer (safe: distinct buffer)
        if (kt + 1 < NKT) {
            v8bf a;
#pragma unroll
            for (int j = 0; j < 4; ++j) { a[j] = (bf16_t)(f0[j] * aw); a[4 + j] = (bf16_t)(f1[j] * aw); }
            *(v8bf*)(&sA[nxt * MT * AROW + r0 * AROW + c0]) = a;
        }

        asm volatile("s_wait_asynccnt 0" ::: "memory");
        __syncthreads();
    }

    // ---- epilogue: +b1, ReLU, tiny GEMM vs W2 [H][2], reduce over H ----
    float p0[8], p1[8];
#pragma unroll
    for (int r = 0; r < 8; ++r) { p0[r] = 0.f; p1[r] = 0.f; }

#pragma unroll
    for (int nt = 0; nt < NT; ++nt) {
        int h = hg * 256 + nt * 16 + l16;
        float bb  = b1[h];
        float w20 = W2[h * 2 + 0];
        float w21 = W2[h * 2 + 1];
#pragma unroll
        for (int r = 0; r < 8; ++r) {
            float v = acc[nt][r] + bb;
            v = v > 0.f ? v : 0.f;
            p0[r] += v * w20;
            p1[r] += v * w21;
        }
    }
#pragma unroll
    for (int r = 0; r < 8; ++r) {
        int ml = msub * 16 + (half ? 8 : 0) + r;   // local row (M in C/D layout)
        atomicAdd(&sPL[ml * 2 + 0], p0[r]);
        atomicAdd(&sPL[ml * 2 + 1], p1[r]);
    }
    __syncthreads();

    // ---- final: +b2, write patch_logits, segment-sum into bag logits ----
    if (tid < MT * C_DIM) {
        int ml = tid >> 1, c = tid & 1;
        float v  = sPL[tid] + b2[c];
        int  gm  = m0 + ml;
        if (gm < N) {
            out[NB * C_DIM + gm * C_DIM + c] = v;   // patch_logits after logits
            int accsz = 0, bag = NB - 1;
            for (int b = 0; b < NB; ++b) {
                accsz += bag_sizes[b];
                if (gm < accsz) { bag = b; break; }
            }
            atomicAdd(&out[bag * C_DIM + c], v);
        }
    }
}

// ---------------------------------------------------------------------------
extern "C" void kernel_launch(void* const* d_in, const int* in_sizes, int n_in,
                              void* d_out, int out_size, void* d_ws, size_t ws_size,
                              hipStream_t stream) {
    const float* F   = (const float*)d_in[0];
    const float* att = (const float*)d_in[1];
    const int*   bs  = (const int*)d_in[2];
    const float* W1  = (const float*)d_in[3];
    const float* b1  = (const float*)d_in[4];
    const float* W2  = (const float*)d_in[5];
    const float* b2  = (const float*)d_in[6];
    float* out = (float*)d_out;

    int NB = in_sizes[2];                 // 64 bags
    int H  = in_sizes[4];                 // 512
    int Dd = in_sizes[3] / H;             // 1024
    int N  = in_sizes[0] / Dd;            // 131072
    (void)n_in; (void)out_size; (void)ws_size;

    bf16_t* w1p = (bf16_t*)d_ws;          // 1 MB bf16 packed W1

    int packN = D_DIM * H_DIM;
    pack_w1_kernel<<<(packN + 255) / 256, 256, 0, stream>>>(W1, w1p, out, NB * C_DIM);
    mil_fused_kernel<<<N / MT, 256, 0, stream>>>(F, att, bs, w1p, b1, W2, b2, out, N, NB);
}
